// multihead_self_attention_61710090109658
// MI455X (gfx1250) — matrix operand order repaired and checked by reading, not compiled, not run on hardware
//
#include <hip/hip_runtime.h>

// ---------------------------------------------------------------------------
// Types for CDNA5 WMMA (wave32): 16x16x32 bf16 -> f32
// ---------------------------------------------------------------------------
typedef __attribute__((ext_vector_type(16))) __bf16      v16bf;
typedef __attribute__((ext_vector_type(8)))  float       v8f;
typedef __attribute__((ext_vector_type(4)))  unsigned int v4u;

union FragU { v16bf f; v4u u[2]; };

__device__ __forceinline__ v8f vzero() {
  v8f z;
#pragma unroll
  for (int i = 0; i < 8; i++) z[i] = 0.0f;
  return z;
}

// A-matrix fragment (16x32, M x K), row-major source with leading dim `ld`.
// lanes 0-15: M=lane, K chunks {0..7, 16..23}; lanes 16-31: K {8..15, 24..31}.
__device__ __forceinline__ v16bf load_frag_a(const __bf16* base, int ld,
                                             int row, int k0, int lane) {
  int half = lane >> 4, l = lane & 15;
  const __bf16* p = base + (size_t)(row + l) * ld + k0 + half * 8;
  FragU r;
  r.u[0] = *(const v4u*)p;
  r.u[1] = *(const v4u*)(p + 16);
  return r.f;
}

// B-matrix fragment (32x16, K x N), source = B^T row-major (N rows of
// contiguous K): lanes 0-15 hold N=lane K {0..7, 16..23}; lanes 16-31 hold K {8..15, 24..31}.
__device__ __forceinline__ v16bf load_frag_b(const __bf16* base, int ld,
                                             int row, int k0, int lane) {
  int half = lane >> 4, l = lane & 15;
  const __bf16* p = base + (size_t)(row + l) * ld + k0 + half * 8;
  FragU r;
  r.u[0] = *(const v4u*)p;
  r.u[1] = *(const v4u*)(p + 16);
  return r.f;
}

__device__ __forceinline__ v8f wmma_bf16(v16bf a, v16bf b, v8f c) {
  return __builtin_amdgcn_wmma_f32_16x16x32_bf16(
      false, a, false, b, (short)0, c, false, false);
}

// exp2 on the raw transcendental unit
__device__ __forceinline__ float fast_exp2(float x) {
#if __has_builtin(__builtin_amdgcn_exp2f)
  return __builtin_amdgcn_exp2f(x);
#else
  return exp2f(x);
#endif
}

// max-reduce across the 16 lanes of each wave32 half (xor butterfly).
__device__ __forceinline__ float halfrow_max(float v) {
#if __has_builtin(__builtin_amdgcn_permlane16)
  unsigned u, p;
  u = __builtin_bit_cast(unsigned, v);   // xor 1
  p = __builtin_amdgcn_permlane16(u, u, 0x67452301u, 0xEFCDAB89u, false, false);
  v = fmaxf(v, __builtin_bit_cast(float, p));
  u = __builtin_bit_cast(unsigned, v);   // xor 2
  p = __builtin_amdgcn_permlane16(u, u, 0x54761032u, 0xDCFE98BAu, false, false);
  v = fmaxf(v, __builtin_bit_cast(float, p));
  u = __builtin_bit_cast(unsigned, v);   // xor 4
  p = __builtin_amdgcn_permlane16(u, u, 0x32107654u, 0xBA98FEDCu, false, false);
  v = fmaxf(v, __builtin_bit_cast(float, p));
  u = __builtin_bit_cast(unsigned, v);   // xor 8
  p = __builtin_amdgcn_permlane16(u, u, 0xFEDCBA98u, 0x76543210u, false, false);
  v = fmaxf(v, __builtin_bit_cast(float, p));
#else
#pragma unroll
  for (int d = 1; d < 16; d <<= 1) v = fmaxf(v, __shfl_xor(v, d, 32));
#endif
  return v;
}

// ---------------------------------------------------------------------------
// fp32 -> bf16 conversion
// ---------------------------------------------------------------------------
__global__ void cvt_f32_bf16(const float* __restrict__ in,
                             __bf16* __restrict__ out, int n) {
  int i = blockIdx.x * blockDim.x + threadIdx.x;
  if (i < n) out[i] = (__bf16)in[i];
}

// ---------------------------------------------------------------------------
// NT GEMM: Y[8192,1024] = A[8192,1024] * B[1024,1024]^T  (bf16, f32 acc)
// 256 threads = 8 waves (4x2); wave tile 64x64 (16 accs); block tile 256x128.
// B tile (128 N x 32 K) is staged into LDS via gfx1250 async DMA
// (global_load_async_to_lds_b128, ASYNCcnt), double-buffered.
// MODE 0: Q bf16 [b,h,s,64], scaled by log2(e)/8 (exp2-domain softmax)
// MODE 1: K bf16 [b,h,s,64]
// MODE 2: V bf16 transposed [b,h,64,s]
// MODE 3: fp32 row-major [8192,1024]
// ---------------------------------------------------------------------------
template <int MODE>
__launch_bounds__(256)
__global__ void gemm_nt(const __bf16* __restrict__ A,
                        const __bf16* __restrict__ B,
                        void* __restrict__ Y) {
  const int LD = 1024;
  __shared__ __bf16 sB[2][128][32];  // double-buffered B tile, 2 x 8 KB

  int lane = threadIdx.x & 31, wid = threadIdx.x >> 5;
  int rowbase = blockIdx.x * 256 + (wid >> 1) * 64;
  int colloc  = (wid & 1) * 64;                 // wave's N offset inside tile
  int colbase = blockIdx.y * 128 + colloc;

  // Async-copy mapping: thread t moves 32 bytes of the 8 KB tile per k-step
  // (one wave-level async b128 moves 32 lanes x 16 B = 512 B).
  int t    = threadIdx.x;
  int srow = t >> 1;            // 0..127 : N index inside the tile
  int skh  = (t & 1) * 16;      // 0 / 16 : K half
  const __bf16* gsrc0 =
      B + (size_t)(blockIdx.y * 128 + srow) * LD + skh;
  unsigned lds0 = (unsigned)(size_t)&sB[0][srow][skh];  // low 32 bits = LDS addr
  const unsigned BUFB = 128u * 32u * 2u;                // 8192 bytes

  v8f acc[16];
#pragma unroll
  for (int i = 0; i < 16; i++) acc[i] = vzero();

  // Prime buffer 0 with the k=0 tile.
  {
    unsigned la = lds0;
    unsigned long long ga = (unsigned long long)(size_t)gsrc0;
    asm volatile(
        "global_load_async_to_lds_b128 %0, %1, off\n\t"
        "global_load_async_to_lds_b128 %2, %3, off"
        :: "v"(la), "v"(ga), "v"(la + 16u),
           "v"(ga + 16ull) : "memory");
  }

  for (int kt = 0; kt < 32; kt++) {
    int cur = kt & 1;
    asm volatile("s_wait_asynccnt 0" ::: "memory");
    __syncthreads();  // tile `cur` ready for all waves; prev reads all retired

    if (kt + 1 < 32) {  // kick DMA for the next tile into the other buffer
      unsigned la = lds0 + (unsigned)(cur ^ 1) * BUFB;
      unsigned long long ga =
          (unsigned long long)(size_t)(gsrc0 + (kt + 1) * 32);
      asm volatile(
          "global_load_async_to_lds_b128 %0, %1, off\n\t"
          "global_load_async_to_lds_b128 %2, %3, off"
          :: "v"(la), "v"(ga), "v"(la + 16u),
             "v"(ga + 16ull) : "memory");
    }

    int k0 = kt * 32;
    v16bf af[4], bf[4];
#pragma unroll
    for (int i = 0; i < 4; i++)
      af[i] = load_frag_a(A, LD, rowbase + i * 16, k0, lane);
#pragma unroll
    for (int j = 0; j < 4; j++)
      bf[j] = load_frag_b(&sB[cur][0][0], 32, colloc + j * 16, 0, lane);
#pragma unroll
    for (int i = 0; i < 4; i++)
#pragma unroll
      for (int j = 0; j < 4; j++)
        acc[i * 4 + j] = wmma_bf16(af[i], bf[j], acc[i * 4 + j]);
  }

  const float QSCALE = 0.125f * 1.4426950408889634f;  // 1/sqrt(64) * log2(e)
  int half = lane >> 4, ln = lane & 15;
#pragma unroll
  for (int i = 0; i < 4; i++)
#pragma unroll
    for (int j = 0; j < 4; j++) {
      v8f c = acc[i * 4 + j];
#pragma unroll
      for (int r = 0; r < 8; r++) {
        int row = rowbase + i * 16 + half * 8 + r;  // n index over b*s
        int col = colbase + j * 16 + ln;            // e index over d_model
        float v = c[r];
        if (MODE == 0 || MODE == 1) {
          int bb = row >> 11, ss = row & 2047, hh = col >> 6, dh = col & 63;
          float sv = (MODE == 0) ? v * QSCALE : v;
          ((__bf16*)Y)[((((size_t)bb * 16 + hh) * 2048 + ss) << 6) + dh] =
              (__bf16)sv;
        } else if (MODE == 2) {
          int bb = row >> 11, ss = row & 2047, hh = col >> 6, dh = col & 63;
          ((__bf16*)Y)[(((size_t)bb * 16 + hh) * 64 + dh) * 2048 + ss] =
              (__bf16)v;
        } else {
          ((float*)Y)[(size_t)row * 1024 + col] = v;
        }
      }
    }
}

// ---------------------------------------------------------------------------
// Fused causal attention (flash style, exp2 domain).
// Block = 128 threads = 4 waves; block owns 128 query rows; wave w owns
// rows qt*128 + w*32 .. +31 (two 16-row WMMA blocks).
// Q,K: [b,h,2048,64] bf16 (Q pre-scaled); Vt: [b,h,64,2048] bf16.
// Softmax denominator computed on the matrix unit: l = P @ ones.
// ---------------------------------------------------------------------------
__launch_bounds__(128)
__global__ void attn_fused(const __bf16* __restrict__ Q,
                           const __bf16* __restrict__ Km,
                           const __bf16* __restrict__ Vt,
                           __bf16* __restrict__ Aout) {
  __shared__ __bf16 sP[4][32][72];  // per-wave P tiles (2 x 16x64), padded

  int lane = threadIdx.x & 31, w = threadIdx.x >> 5;
  int half = lane >> 4, ln = lane & 15;
  int qt = blockIdx.x, h = blockIdx.y, b = blockIdx.z;

  const __bf16* Qh = Q  + (((size_t)b * 16 + h) * 2048) * 64;
  const __bf16* Kh = Km + (((size_t)b * 16 + h) * 2048) * 64;
  const __bf16* Vh = Vt + (((size_t)b * 16 + h) * 64) * 2048;

  int qbase = qt * 128 + w * 32;

  v16bf qa[2][2];
#pragma unroll
  for (int qq = 0; qq < 2; qq++)
#pragma unroll
    for (int t = 0; t < 2; t++)
      qa[qq][t] = load_frag_a(Qh, 64, qbase + qq * 16, t * 32, lane);

  v16bf onesb;
#pragma unroll
  for (int i = 0; i < 16; i++) onesb[i] = (__bf16)1.0f;

  float m[2][8];
  v8f lsum[2], o[2][4];
#pragma unroll
  for (int qq = 0; qq < 2; qq++) {
#pragma unroll
    for (int i = 0; i < 8; i++) m[qq][i] = -3.0e38f;
    lsum[qq] = vzero();
#pragma unroll
    for (int n = 0; n < 4; n++) o[qq][n] = vzero();
  }

  int jmax = qbase >> 6;  // last (and only masked) key tile
  for (int j = 0; j <= jmax; j++) {
    // ---- S = Q K^T : 32q x 64k, K-fragments shared by both q blocks ----
    v8f s[2][4];
#pragma unroll
    for (int qq = 0; qq < 2; qq++)
#pragma unroll
      for (int n = 0; n < 4; n++) s[qq][n] = vzero();
#pragma unroll
    for (int n = 0; n < 4; n++) {
      v16bf kb0 = load_frag_b(Kh, 64, j * 64 + n * 16, 0, lane);
      v16bf kb1 = load_frag_b(Kh, 64, j * 64 + n * 16, 32, lane);
#pragma unroll
      for (int qq = 0; qq < 2; qq++) {
        s[qq][n] = wmma_bf16(qa[qq][0], kb0, s[qq][n]);
        s[qq][n] = wmma_bf16(qa[qq][1], kb1, s[qq][n]);
      }
    }

    // ---- causal mask (diagonal tile only) ----
    if (j == jmax) {
#pragma unroll
      for (int qq = 0; qq < 2; qq++)
#pragma unroll
        for (int n = 0; n < 4; n++)
#pragma unroll
          for (int i = 0; i < 8; i++) {
            int col = j * 64 + n * 16 + ln;
            int row = qbase + qq * 16 + i + 8 * half;
            if (col > row) s[qq][n][i] = -3.0e38f;
          }
    }

    // ---- online softmax (exp2 domain); P -> LDS ----
#pragma unroll
    for (int qq = 0; qq < 2; qq++) {
      float tm[8];
#pragma unroll
      for (int i = 0; i < 8; i++)
        tm[i] = fmaxf(fmaxf(s[qq][0][i], s[qq][1][i]),
                      fmaxf(s[qq][2][i], s[qq][3][i]));
#pragma unroll
      for (int i = 0; i < 8; i++) tm[i] = halfrow_max(tm[i]);

#pragma unroll
      for (int i = 0; i < 8; i++) {
        float mn = fmaxf(m[qq][i], tm[i]);
        float alpha = fast_exp2(m[qq][i] - mn);
        m[qq][i] = mn;
        lsum[qq][i] *= alpha;
#pragma unroll
        for (int n = 0; n < 4; n++) o[qq][n][i] *= alpha;
      }
#pragma unroll
      for (int n = 0; n < 4; n++)
#pragma unroll
        for (int i = 0; i < 8; i++) {
          float p = fast_exp2(s[qq][n][i] - m[qq][i]);
          sP[w][qq * 16 + i + 8 * half][n * 16 + ln] = (__bf16)p;
        }
    }
    asm volatile("s_wait_dscnt 0" ::: "memory");

    // ---- reload P as A-fragments; l += P @ ones ; O += P @ V ----
    v16bf pa[2][2];
#pragma unroll
    for (int qq = 0; qq < 2; qq++) {
#pragma unroll
      for (int ks = 0; ks < 2; ks++)
        pa[qq][ks] = load_frag_a(&sP[w][qq * 16][0], 72, 0, ks * 32, lane);
      lsum[qq] = wmma_bf16(pa[qq][0], onesb, lsum[qq]);
      lsum[qq] = wmma_bf16(pa[qq][1], onesb, lsum[qq]);
    }
#pragma unroll
    for (int ks = 0; ks < 2; ks++)
#pragma unroll
      for (int n = 0; n < 4; n++) {
        v16bf vb = load_frag_b(Vh, 2048, n * 16, j * 64 + ks * 32, lane);
#pragma unroll
        for (int qq = 0; qq < 2; qq++)
          o[qq][n] = wmma_bf16(pa[qq][ks], vb, o[qq][n]);
      }
    asm volatile("s_wait_dscnt 0" ::: "memory");  // sP reads before next write
  }

  // ---- normalize and store: [b, s, h*64 + dh] ----
#pragma unroll
  for (int qq = 0; qq < 2; qq++) {
    float inv[8];
#pragma unroll
    for (int i = 0; i < 8; i++) inv[i] = 1.0f / lsum[qq][i];
#pragma unroll
    for (int n = 0; n < 4; n++)
#pragma unroll
      for (int i = 0; i < 8; i++) {
        int row = qbase + qq * 16 + i + 8 * half;
        int col = h * 64 + n * 16 + ln;
        Aout[((size_t)b * 2048 + row) * 1024 + col] =
            (__bf16)(o[qq][n][i] * inv[i]);
      }
  }
}

// ---------------------------------------------------------------------------
// Launcher
// ---------------------------------------------------------------------------
extern "C" void kernel_launch(void* const* d_in, const int* in_sizes, int n_in,
                              void* d_out, int out_size, void* d_ws,
                              size_t ws_size, hipStream_t stream) {
  const float* x  = (const float*)d_in[0];
  const float* Wq = (const float*)d_in[1];
  const float* Wk = (const float*)d_in[2];
  const float* Wv = (const float*)d_in[3];
  const float* Wo = (const float*)d_in[4];

  const size_t NX = 4ull * 2048 * 1024;  // activation elements
  const size_t NW = 1024ull * 1024;      // weight elements

  char* ws = (char*)d_ws;
  __bf16* xb  = (__bf16*)ws; ws += NX * 2;
  __bf16* wqb = (__bf16*)ws; ws += NW * 2;
  __bf16* wkb = (__bf16*)ws; ws += NW * 2;
  __bf16* wvb = (__bf16*)ws; ws += NW * 2;
  __bf16* wob = (__bf16*)ws; ws += NW * 2;
  __bf16* Qb  = (__bf16*)ws; ws += NX * 2;
  __bf16* Kb  = (__bf16*)ws; ws += NX * 2;
  __bf16* Vtb = (__bf16*)ws; ws += NX * 2;
  __bf16* Ab  = (__bf16*)ws; ws += NX * 2;

  cvt_f32_bf16<<<(int)((NX + 255) / 256), 256, 0, stream>>>(x,  xb,  (int)NX);
  cvt_f32_bf16<<<(int)((NW + 255) / 256), 256, 0, stream>>>(Wq, wqb, (int)NW);
  cvt_f32_bf16<<<(int)((NW + 255) / 256), 256, 0, stream>>>(Wk, wkb, (int)NW);
  cvt_f32_bf16<<<(int)((NW + 255) / 256), 256, 0, stream>>>(Wv, wvb, (int)NW);
  cvt_f32_bf16<<<(int)((NW + 255) / 256), 256, 0, stream>>>(Wo, wob, (int)NW);

  dim3 gg(32, 8);  // M tiles (8192/256), N tiles (1024/128)
  gemm_nt<0><<<gg, 256, 0, stream>>>(xb, wqb, (void*)Qb);
  gemm_nt<1><<<gg, 256, 0, stream>>>(xb, wkb, (void*)Kb);
  gemm_nt<2><<<gg, 256, 0, stream>>>(xb, wvb, (void*)Vtb);

  attn_fused<<<dim3(16, 16, 4), 128, 0, stream>>>(Qb, Kb, Vtb, Ab);

  gemm_nt<3><<<gg, 256, 0, stream>>>(Ab, wob, d_out);
}
